// GeometryEncoder_83803401880437
// MI455X (gfx1250) — matrix operand order repaired and checked by reading, not compiled, not run on hardware
//
#include <hip/hip_runtime.h>
#include <hip/hip_bf16.h>
#include <math.h>

#define KMAX_PTS 30          // PMAX
#define NUM_PTS 5
#define C1 112               // conv1 out channels
#define C2 224               // conv2 out channels
#define CPAD 128             // padded conv1 channels for clean K=32 WMMA steps
#define PE_DIM 32
#define GEO_DIM 256
#define POLY_PER_BLK 16
#define ROWS_PER_BLK (POLY_PER_BLK * NUM_PTS)   // 80
#define M_TILES 5            // 80 / 16
#define N_TILES 14           // 224 / 16
#define K_STEPS 12           // 3 taps * 128 ch / 32

typedef __attribute__((ext_vector_type(16))) _Float16 v16h;
typedef __attribute__((ext_vector_type(8)))  _Float16 v8h;
typedef __attribute__((ext_vector_type(8)))  float    v8f;

// ---------------------------------------------------------------------------
// Kernel 0: pack w2 [224][112][3] into WMMA B fragments (f16), im2col layout:
//   reduction index k = tap*128 + c  (c in [0,128), zeros for c>=112)
//   fragment id = nt*K_STEPS*32 + ks*32 + lane ; 16 halves per fragment,
//   half j = B[kbase + j, n]  with n = nt*16 + (lane&15),
//   kbase = ks*32 + (lane>>4)*16   (VGPR i packs K=2i,2i+1)
// ---------------------------------------------------------------------------
__global__ void pack_w2_kernel(const float* __restrict__ w2, _Float16* __restrict__ wsB) {
    int idx = blockIdx.x * 256 + threadIdx.x;
    if (idx >= N_TILES * K_STEPS * 32) return;
    int lane = idx & 31;
    int ks   = (idx >> 5) % K_STEPS;
    int nt   = idx / (32 * K_STEPS);
    int n     = nt * 16 + (lane & 15);
    int kbase = ks * 32 + (lane >> 4) * 16;
    _Float16* dst = wsB + (size_t)idx * 16;
#pragma unroll
    for (int j = 0; j < 16; ++j) {
        int kk = kbase + j;
        int t  = kk >> 7;          // tap 0..2
        int c  = kk & 127;         // channel
        float v = (c < C1) ? w2[((size_t)n * C1 + c) * 3 + t] : 0.0f;
        dst[j] = (_Float16)v;
    }
}

// ---------------------------------------------------------------------------
// Kernel 1: per-polyline arc-length resample + sinusoidal PE.
// One thread per polyline. Writes PE into feats[:, :, 0:32] and normalized
// coords into coordsOut [K,5,2] (== second reference output).
// ---------------------------------------------------------------------------
__global__ void resample_pe_kernel(const float* __restrict__ geoms,
                                   const int*   __restrict__ lengths,
                                   const int*   __restrict__ roi_w,
                                   const int*   __restrict__ roi_h,
                                   float* __restrict__ feats,
                                   float* __restrict__ coordsOut,
                                   int K) {
    int k = blockIdx.x * blockDim.x + threadIdx.x;
    if (k >= K) return;
    const float* g = geoms + (size_t)k * KMAX_PTS * 2;
    int L = lengths[k];                       // >= 2

    // pass 1: total masked arc length (same summation order as pass 2)
    float p0x = g[0], p0y = g[1];
    float px = p0x, py = p0y;
    float total = 0.0f;
    for (int m = 1; m < KMAX_PTS; ++m) {
        float cx = g[2 * m], cy = g[2 * m + 1];
        if (m < L) {
            float dx = cx - px, dy = cy - py;
            float sq = dx * dx + dy * dy;
            if (sq > 0.0f) total += sqrtf(sq);
        }
        px = cx; py = cy;
    }

    float rx[NUM_PTS], ry[NUM_PTS];
    if (total < 1e-6f) {
#pragma unroll
        for (int j = 0; j < NUM_PTS; ++j) { rx[j] = p0x; ry[j] = p0y; }
    } else {
        float targ[NUM_PTS];
#pragma unroll
        for (int j = 0; j < NUM_PTS; ++j) targ[j] = 0.25f * (float)j * total;
        // pass 2: walk segments; first m with cum[m] >= targ == searchsorted-left
        float cum = 0.0f;
        int j = 0;
        px = p0x; py = p0y;
        float lx = p0x, ly = p0y;
        for (int m = 1; m < KMAX_PTS && j < NUM_PTS; ++m) {
            float cx = g[2 * m], cy = g[2 * m + 1];
            float dx = cx - px, dy = cy - py;
            float s = 0.0f;
            if (m < L) {
                float sq = dx * dx + dy * dy;
                if (sq > 0.0f) s = sqrtf(sq);
            }
            float cn = cum + s;
            while (j < NUM_PTS && cn >= targ[j]) {
                float t = (targ[j] - cum) / (s + 1e-8f);
                rx[j] = px + t * dx;
                ry[j] = py + t * dy;
                ++j;
            }
            cum = cn; px = cx; py = cy; lx = cx; ly = cy;
        }
        for (; j < NUM_PTS; ++j) { rx[j] = lx; ry[j] = ly; }  // fp safety net
    }

    float fw = (float)roi_w[0], fh = (float)roi_h[0];
    float hw = 0.5f * fw, hh = 0.5f * fh;
    const float PI = 3.14159265358979323846f;
#pragma unroll
    for (int j = 0; j < NUM_PTS; ++j) {
        float nx = (rx[j] + hw) / fw;
        float ny = (ry[j] + hh) / fh;
        float2* cp = (float2*)(coordsOut + ((size_t)k * NUM_PTS + j) * 2);
        *cp = make_float2(nx, ny);
        float buf[PE_DIM];
#pragma unroll
        for (int f = 0; f < 8; ++f) {
            float fr = (float)(1 << f) * PI;
            float ax = nx * fr, ay = ny * fr;
            buf[f]      = sinf(ax);
            buf[8 + f]  = cosf(ax);
            buf[16 + f] = sinf(ay);
            buf[24 + f] = cosf(ay);
        }
        float4* o4 = (float4*)(feats + ((size_t)k * NUM_PTS + j) * GEO_DIM);
#pragma unroll
        for (int q = 0; q < 8; ++q)
            o4[q] = make_float4(buf[4 * q], buf[4 * q + 1], buf[4 * q + 2], buf[4 * q + 3]);
    }
}

// ---------------------------------------------------------------------------
// Kernel 2: conv1 (VALU, to padded f16 LDS) -> conv2 via v_wmma_f32_16x16x32_f16
// -> bias + ReLU -> LayerNorm(224) -> feats[:, :, 32:256].
// 16 polylines / block, 256 threads (8 wave32). B fragments loaded once per
// nt and reused across all 5 M-tiles; mt loop is NOT unrolled so A-fragment
// ds_loads stay inside one tile's lifetime (keeps VGPRs < 256, no spills).
// ---------------------------------------------------------------------------
__global__ __launch_bounds__(256) void conv_ln_kernel(
        const float*   __restrict__ coordsBuf,   // [K,5,2] normalized
        const _Float16* __restrict__ wsB,        // packed B fragments
        const float* __restrict__ w1,            // [112][2][3]
        const float* __restrict__ b1,            // [112]
        const float* __restrict__ b2,            // [224]
        const float* __restrict__ gamma,         // [224]
        const float* __restrict__ beta,          // [224]
        float* __restrict__ feats, int K) {
    __shared__ _Float16 sX[POLY_PER_BLK * 7 * CPAD];   // padded conv1 out (28672 B)
    __shared__ float sY[ROWS_PER_BLK * C2];            // conv2 relu out   (71680 B)
    __shared__ float sW1[C1 * 6];
    __shared__ float sB1[C1];
    __shared__ float sB2[C2];
    __shared__ float sG[C2];
    __shared__ float sBt[C2];
    __shared__ float sC[POLY_PER_BLK * NUM_PTS * 2];   // coords (160 floats)
    __shared__ float sMu[ROWS_PER_BLK];
    __shared__ float sRs[ROWS_PER_BLK];

    const int tid = threadIdx.x;
    const int blk = blockIdx.x;
    const int polyBase = blk * POLY_PER_BLK;

    // ---- stage parameters & coords into LDS, zero padded X tile ----
    for (int i = tid; i < C1 * 6; i += 256) sW1[i] = w1[i];
    for (int i = tid; i < C1; i += 256)     sB1[i] = b1[i];
    for (int i = tid; i < C2; i += 256) { sB2[i] = b2[i]; sG[i] = gamma[i]; sBt[i] = beta[i]; }
    for (int i = tid; i < POLY_PER_BLK * NUM_PTS * 2; i += 256) {
        int gp = polyBase + i / (NUM_PTS * 2);
        sC[i] = (gp < K) ? coordsBuf[(size_t)polyBase * NUM_PTS * 2 + i] : 0.0f;
    }
    unsigned* zx = (unsigned*)sX;
    for (int i = tid; i < POLY_PER_BLK * 7 * CPAD / 2; i += 256) zx[i] = 0u;
    __syncthreads();

    // ---- conv1 (2 -> 112, k=3, same pad) + ReLU -> f16 padded LDS ----
    for (int i = tid; i < POLY_PER_BLK * NUM_PTS * C1; i += 256) {
        int c    = i % C1;
        int pp   = i / C1;
        int pos  = pp % NUM_PTS;
        int poly = pp / NUM_PTS;
        float acc = sB1[c];
#pragma unroll
        for (int t = 0; t < 3; ++t) {
            int ip = pos + t - 1;
            if (ip >= 0 && ip < NUM_PTS) {
                acc += sW1[c * 6 + t]     * sC[poly * 10 + ip * 2];
                acc += sW1[c * 6 + 3 + t] * sC[poly * 10 + ip * 2 + 1];
            }
        }
        acc = fmaxf(acc, 0.0f);
        sX[poly * (7 * CPAD) + (pos + 1) * CPAD + c] = (_Float16)acc;
    }
    __syncthreads();

    // ---- conv2 as WMMA GEMM: 80x224, reduction 3*128 in 12 K=32 steps ----
    const int wave = tid >> 5;
    const int lane = tid & 31;
    const int r    = lane & 15;         // M within tile
    const int kh   = (lane >> 4) * 16;  // A half-group select
    for (int nt = wave; nt < N_TILES; nt += 8) {
        // load this wave's 12 B fragments once; reuse across all 5 M-tiles
        const _Float16* bptr = wsB + ((size_t)nt * K_STEPS * 32 + lane) * 16;
        v16h bf[K_STEPS];
#pragma unroll
        for (int ks = 0; ks < K_STEPS; ++ks)
            bf[ks] = *(const v16h*)(bptr + (size_t)ks * 32 * 16);
        const int n  = lane & 15;
        const int ch = nt * 16 + n;
        const float bias = sB2[ch];
#pragma unroll 1   // keep A loads inside one tile: bounds VGPR pressure, no spills
        for (int mt = 0; mt < M_TILES; ++mt) {
            int row  = mt * 16 + r;
            int poly = row / NUM_PTS;
            int pos  = row % NUM_PTS;
            v8f acc = {};
#pragma unroll
            for (int ks = 0; ks < K_STEPS; ++ks) {
                int tap = ks >> 2;               // which conv tap (0..2)
                int cb  = (ks & 3) * 32;         // channel base within tap
                int o1  = poly * (7 * CPAD) + (pos + tap) * CPAD + cb + kh;
                v8h lo = *(const v8h*)(&sX[o1]);
                v8h hi = *(const v8h*)(&sX[o1 + 8]);
                v16h a;
#pragma unroll
                for (int j = 0; j < 8; ++j) { a[j] = lo[j]; a[8 + j] = hi[j]; }
                acc = __builtin_amdgcn_wmma_f32_16x16x32_f16(
                          false, a, false, bf[ks], (short)0, acc, false, false);
            }
            // scatter C tile (VGPR v <-> M = v or v+8; lane%16 <-> N) + bias + ReLU
#pragma unroll
            for (int v = 0; v < 8; ++v) {
                int m = (lane < 16) ? v : (v + 8);
                sY[(mt * 16 + m) * C2 + ch] = fmaxf(acc[v] + bias, 0.0f);
            }
        }
    }
    __syncthreads();

    // ---- LayerNorm stats per row (float4 reads) ----
    if (tid < ROWS_PER_BLK) {
        const float4* rp = (const float4*)&sY[tid * C2];
        float s = 0.0f;
        for (int c = 0; c < C2 / 4; ++c) {
            float4 y = rp[c];
            s += y.x + y.y + y.z + y.w;
        }
        float mu = s * (1.0f / C2);
        float v = 0.0f;
        for (int c = 0; c < C2 / 4; ++c) {
            float4 y = rp[c];
            float d0 = y.x - mu, d1 = y.y - mu, d2 = y.z - mu, d3 = y.w - mu;
            v += d0 * d0 + d1 * d1 + d2 * d2 + d3 * d3;
        }
        v *= (1.0f / C2);
        sMu[tid] = mu;
        sRs[tid] = rsqrtf(v + 1e-5f);
    }
    __syncthreads();

    // ---- normalize + affine + vectorized store to feats[:, :, 32:256] ----
    const float4* sY4 = (const float4*)sY;
    for (int i = tid; i < ROWS_PER_BLK * (C2 / 4); i += 256) {
        int rowi = i / (C2 / 4);
        int c4   = (i % (C2 / 4)) * 4;
        int poly = polyBase + rowi / NUM_PTS;
        if (poly >= K) continue;
        int pos  = rowi % NUM_PTS;
        float mu = sMu[rowi], rs = sRs[rowi];
        float4 y = sY4[i];
        float4 o;
        o.x = (y.x - mu) * rs * sG[c4]     + sBt[c4];
        o.y = (y.y - mu) * rs * sG[c4 + 1] + sBt[c4 + 1];
        o.z = (y.z - mu) * rs * sG[c4 + 2] + sBt[c4 + 2];
        o.w = (y.w - mu) * rs * sG[c4 + 3] + sBt[c4 + 3];
        *(float4*)(feats + ((size_t)poly * NUM_PTS + pos) * GEO_DIM + PE_DIM + c4) = o;
    }
}

// ---------------------------------------------------------------------------
extern "C" void kernel_launch(void* const* d_in, const int* in_sizes, int n_in,
                              void* d_out, int out_size, void* d_ws, size_t ws_size,
                              hipStream_t stream) {
    const float* geoms   = (const float*)d_in[0];
    const int*   lengths = (const int*)  d_in[1];
    const float* w1      = (const float*)d_in[2];
    const float* b1      = (const float*)d_in[3];
    const float* w2      = (const float*)d_in[4];
    const float* b2      = (const float*)d_in[5];
    const float* gamma   = (const float*)d_in[6];
    const float* beta    = (const float*)d_in[7];
    const int*   roi_w   = (const int*)  d_in[8];
    const int*   roi_h   = (const int*)  d_in[9];

    const int K = in_sizes[1];                       // number of polylines
    float* feats     = (float*)d_out;                         // [K,5,256]
    float* coordsOut = feats + (size_t)K * NUM_PTS * GEO_DIM; // [K,5,2]
    _Float16* wsB = (_Float16*)d_ws;                 // 14*12*32*16 halves = 172 KB

    int packThreads = N_TILES * K_STEPS * 32;        // 5376
    pack_w2_kernel<<<(packThreads + 255) / 256, 256, 0, stream>>>(w2, wsB);

    resample_pe_kernel<<<(K + 255) / 256, 256, 0, stream>>>(
        geoms, lengths, roi_w, roi_h, feats, coordsOut, K);

    int nBlk = (K + POLY_PER_BLK - 1) / POLY_PER_BLK;
    conv_ln_kernel<<<nBlk, 256, 0, stream>>>(
        coordsOut, wsB, w1, b1, b2, gamma, beta, feats, K);
}
